// GNN_3100966387952
// MI455X (gfx1250) — matrix operand order repaired and checked by reading, not compiled, not run on hardware
//
#include <hip/hip_runtime.h>

#define H    64
#define NFEA 32
#define RR   4
#define MAXD 10
#define NB   (MAXD + 1)
#define OUTF 16

typedef __attribute__((ext_vector_type(16))) __bf16 v16bf;
typedef __attribute__((ext_vector_type(8)))  float  v8f;
typedef unsigned short bfraw;   // bf16 bit pattern storage

// ---------- TDM availability ----------
#if defined(__has_builtin)
#  if __has_builtin(__builtin_amdgcn_tensor_load_to_lds) && \
      __has_builtin(__builtin_amdgcn_s_wait_tensorcnt)
#    define HAVE_TDM 1
#  endif
#endif
#ifndef HAVE_TDM
#  define HAVE_TDM 0
#endif

// ---------- bf16 helpers (bit-level) ----------
__device__ __forceinline__ bfraw f2bf(float f) {
  union { float f; unsigned u; } c; c.f = f;
  unsigned r = c.u + 0x7FFFu + ((c.u >> 16) & 1u);   // round-to-nearest-even
  return (bfraw)(r >> 16);
}
__device__ __forceinline__ float bf2f(bfraw b) {
  union { unsigned u; float f; } c; c.u = ((unsigned)b) << 16; return c.f;
}

struct Frag { union { v16bf v; uint4 u[2]; }; };

// CDNA5 16-bit A fragment, 16x32 (MxK): lane l holds row (l&15);
// lanes 0-15: K = kh*32 + [0..7] and [16..23]; lanes 16-31: +8.
__device__ __forceinline__ v16bf load_a_frag(const bfraw* A, int row, int kbase) {
  Frag r;
  const bfraw* p = A + row * H + kbase;
  r.u[0] = *reinterpret_cast<const uint4*>(p);        // K = kbase .. +7
  r.u[1] = *reinterpret_cast<const uint4*>(p + 16);   // K = kbase+16 .. +23
  return r.v;
}
// B fragment, 32x16 (KxN) from LDS holding Bt[n][k] (k-contiguous, stride H):
// lane l holds col (l&15); lanes 0-15: K = kh*32 + [0..7] and [16..23]; lanes 16-31: +8.
__device__ __forceinline__ v16bf load_b_frag(const bfraw* Bt, int col, int g, int kh) {
  Frag r;
  const bfraw* p = Bt + col * H + kh * 32 + g * 8;
  r.u[0] = *reinterpret_cast<const uint4*>(p);
  r.u[1] = *reinterpret_cast<const uint4*>(p + 16);
  return r.v;
}
__device__ __forceinline__ v8f wmma_bf(v16bf a, v16bf b, v8f c) {
  return __builtin_amdgcn_wmma_f32_16x16x32_bf16(false, a, false, b, (short)0, c, false, false);
}

// ---------- LDS staging: TDM (tensor DMA) with b128 fallback ----------
#if HAVE_TDM
typedef unsigned int u32x4 __attribute__((ext_vector_type(4)));
typedef int i32x4 __attribute__((ext_vector_type(4)));
typedef int i32x8 __attribute__((ext_vector_type(8)));
typedef __attribute__((address_space(3))) unsigned short* lds_us_p;

// 1-D tile of `nelem` 2-byte elements: global -> LDS via Tensor Data Mover.
__device__ __forceinline__ void tdm_load_2b(const bfraw* gsrc, bfraw* lds_dst,
                                            unsigned nelem) {
  unsigned long long ga = (unsigned long long)gsrc;
  unsigned lof = (unsigned)(unsigned long long)(lds_us_p)lds_dst;
  u32x4 g0;
  g0[0] = 1u;                                       // count=1, user mode
  g0[1] = lof;                                      // lds_addr
  g0[2] = (unsigned)ga;                             // global_addr[31:0]
  g0[3] = (unsigned)(ga >> 32) | 0x80000000u;       // addr[56:32] | type=2
  i32x8 g1;
  g1[0] = 0x00010000;                               // data_size=1 (2 bytes)
  g1[1] = (int)((nelem & 0xFFFFu) << 16);           // tensor_dim0 lo16
  g1[2] = (int)((nelem >> 16) | (1u << 16));        // tensor_dim0 hi16 | dim1=1
  g1[3] = (int)((nelem & 0xFFFFu) << 16);           // tile_dim0
  g1[4] = 1;                                        // tile_dim1=1
  g1[5] = (int)nelem;                               // tensor_dim0_stride lo32
  g1[6] = 0;
  g1[7] = 0;
  i32x4 z = {0, 0, 0, 0};
#if __clang_major__ >= 23
  i32x8 z8 = {0, 0, 0, 0, 0, 0, 0, 0};
  __builtin_amdgcn_tensor_load_to_lds(g0, g1, z, z, z8, 0);
#else
  __builtin_amdgcn_tensor_load_to_lds(g0, g1, z, z, 0);
#endif
}
#endif

// Stage nelem bf16 from (pre-transposed) global into LDS. Call stage_wait()
// then __syncthreads() before use.
__device__ __forceinline__ void stage_lds(const bfraw* gsrc, bfraw* lds_dst,
                                          int nelem, int tid, int nthreads) {
#if HAVE_TDM
  if (tid < 32) tdm_load_2b(gsrc, lds_dst, (unsigned)nelem);  // wave 0 issues DMA
#else
  for (int i = tid; i < (nelem >> 3); i += nthreads)
    *((uint4*)lds_dst + i) = *((const uint4*)gsrc + i);
#endif
}
__device__ __forceinline__ void stage_wait() {
#if HAVE_TDM
  __builtin_amdgcn_s_wait_tensorcnt(0);
#endif
}

// ---------- elementwise / setup kernels ----------
__global__ void k_zero_f(float* p, int n) {
  int i = blockIdx.x * blockDim.x + threadIdx.x;
  if (i < n) p[i] = 0.0f;
}
__global__ void k_zero_i(int* p, int n) {
  int i = blockIdx.x * blockDim.x + threadIdx.x;
  if (i < n) p[i] = 0;
}
__global__ void k_bf16(const float* in, bfraw* out, int n) {
  int i = blockIdx.x * blockDim.x + threadIdx.x;
  if (i < n) out[i] = f2bf(in[i]);
}
// float [m][K][Nc] row-major  ->  bf16 [m][Nc][K] (per-matrix transpose)
__global__ void k_bf16t(const float* in, bfraw* out, int K, int Nc, int total) {
  int i = blockIdx.x * blockDim.x + threadIdx.x;
  if (i >= total) return;
  int per = K * Nc;
  int m = i / per, r = i % per;
  int k = r / Nc, n = r % Nc;
  out[m * per + n * K + k] = f2bf(in[i]);
}
__global__ void k_relu_dual(const float* in, float* outf, bfraw* outb, int n) {
  int i = blockIdx.x * blockDim.x + threadIdx.x;
  if (i >= n) return;
  float v = fmaxf(in[i], 0.0f);
  outf[i] = v;
  outb[i] = f2bf(v);
}
__global__ void k_embed(const float* x, const float* emb, float* h, int n) {
  int i = blockIdx.x * blockDim.x + threadIdx.x;
  if (i >= n) return;
  const float* xr = x + (size_t)i * NFEA;
  int best = 0; float bv = xr[0];
  for (int j = 1; j < NFEA; ++j) { float v = xr[j]; if (v > bv) { bv = v; best = j; } }
  const float* er = emb + best * H;
  float* hr = h + (size_t)i * H;
  for (int j = 0; j < H; ++j) hr[j] = er[j];
}
__global__ void k_rel(const float* eattr, int* rel, int e) {
  int i = blockIdx.x * blockDim.x + threadIdx.x;
  if (i >= e) return;
  const float* a = eattr + (size_t)i * RR;
  int best = 0; float bv = a[0];
  for (int j = 1; j < RR; ++j) { float v = a[j]; if (v > bv) { bv = v; best = j; } }
  rel[i] = best;
}

// ---------- RGCN scatter path ----------
__global__ void k_cnt(const int* tgt, const int* rel, float* cnt, int e) {
  int i = blockIdx.x * blockDim.x + threadIdx.x;
  if (i >= e) return;
  atomicAdd(&cnt[tgt[i] * RR + rel[i]], 1.0f);
}
__global__ void k_inv(float* cnt, int n) {
  int i = blockIdx.x * blockDim.x + threadIdx.x;
  if (i >= n) return;
  float c = cnt[i];
  cnt[i] = (c > 0.0f) ? (1.0f / c) : 0.0f;
}
// one thread = one edge x 4 features (8B bf16 payload, 4 atomics)
__global__ void k_rgcn_scatter(const bfraw* hx, const float* inv,
                               const int* src, const int* tgt, const int* rel,
                               float* agg, int e, int n) {
  int idx = blockIdx.x * blockDim.x + threadIdx.x;
  if (idx >= e * (H / 4)) return;
  int ed = idx >> 4, f = (idx & 15) * 4;
  int r = rel[ed], s = src[ed], t = tgt[ed];
  float w = inv[t * RR + r];
  const unsigned* hp = (const unsigned*)(hx + ((size_t)r * n + s) * H + f);
  unsigned w0 = hp[0], w1 = hp[1];
  float* ap = agg + (size_t)t * H + f;
  atomicAdd(ap + 0, bf2f((bfraw)(w0 & 0xFFFFu)) * w);
  atomicAdd(ap + 1, bf2f((bfraw)(w0 >> 16)) * w);
  atomicAdd(ap + 2, bf2f((bfraw)(w1 & 0xFFFFu)) * w);
  atomicAdd(ap + 3, bf2f((bfraw)(w1 >> 16)) * w);
}
__global__ void k_combine(const float* agg, const float* hroot, float* h, int n) {
  int i = blockIdx.x * blockDim.x + threadIdx.x;
  if (i < n) h[i] = agg[i] + hroot[i];
}

// ---------- MFConv scatter / bucketing ----------
__global__ void k_deg(const int* tgt, int* deg, int e) {
  int i = blockIdx.x * blockDim.x + threadIdx.x;
  if (i >= e) return;
  atomicAdd(&deg[tgt[i]], 1);
}
// one thread = one edge x 4 features (float4 payload, 4 atomics)
__global__ void k_mf_scatter(const float* hrelu, const int* src, const int* tgt,
                             float* agg, int e) {
  int idx = blockIdx.x * blockDim.x + threadIdx.x;
  if (idx >= e * (H / 4)) return;
  int ed = idx >> 4, f = (idx & 15) * 4;
  float4 v = *(const float4*)(hrelu + (size_t)src[ed] * H + f);
  float* ap = agg + (size_t)tgt[ed] * H + f;
  atomicAdd(ap + 0, v.x);
  atomicAdd(ap + 1, v.y);
  atomicAdd(ap + 2, v.z);
  atomicAdd(ap + 3, v.w);
}
__global__ void k_bucket_count(const int* deg, int* degc, int* bcount, int n) {
  int i = blockIdx.x * blockDim.x + threadIdx.x;
  if (i >= n) return;
  int d = deg[i]; if (d > MAXD) d = MAXD;
  degc[i] = d;
  atomicAdd(&bcount[d], 1);
}
__global__ void k_prefix(const int* bcount, int* boffset, int* btile, int* bfill) {
  if (threadIdx.x == 0 && blockIdx.x == 0) {
    int off = 0, toff = 0;
    for (int d = 0; d < NB; ++d) {
      boffset[d] = off; btile[d] = toff; bfill[d] = 0;
      off += bcount[d];
      toff += (bcount[d] + 15) >> 4;
    }
    btile[NB] = toff;
  }
}
__global__ void k_perm(const int* degc, const int* boffset, int* bfill, int* perm, int n) {
  int i = blockIdx.x * blockDim.x + threadIdx.x;
  if (i >= n) return;
  int d = degc[i];
  int pos = boffset[d] + atomicAdd(&bfill[d], 1);
  perm[pos] = i;
}

// ---------- WMMA GEMM: C[M,64] = A[M,64] @ Bt^T (+bias)(+relu) ----------
// Bt_g: pre-transposed weights, bf16 [64(n)][64(k)], staged to LDS via TDM.
__global__ __launch_bounds__(128) void gemm64(const bfraw* A, const bfraw* Bt_g,
                                              const float* bias, float* Cf, bfraw* Cb,
                                              int M, int relu) {
  __shared__ __align__(16) bfraw Bt[H * H];
  int tid = threadIdx.x;
  stage_lds(Bt_g, Bt, H * H, tid, 128);
  stage_wait();
  __syncthreads();
  int lane = tid & 31, wave = tid >> 5;
  int g = lane >> 4, c0 = lane & 15;
  int rowbase = blockIdx.x * 64 + wave * 16;
  int arow = rowbase + c0; if (arow >= M) arow = M - 1;
  v8f acc[4] = {};
  for (int kh = 0; kh < 2; ++kh) {
    v16bf af = load_a_frag(A, arow, kh * 32 + g * 8);
    for (int ct = 0; ct < 4; ++ct) {
      v16bf bf = load_b_frag(Bt, ct * 16 + c0, g, kh);
      acc[ct] = wmma_bf(af, bf, acc[ct]);
    }
  }
  for (int ct = 0; ct < 4; ++ct) {
    int col = ct * 16 + c0;
    float bv = bias ? bias[col] : 0.0f;
    for (int j = 0; j < 8; ++j) {
      int row = rowbase + j + g * 8;
      if (row < M) {
        float v = acc[ct][j] + bv;
        if (relu) v = fmaxf(v, 0.0f);
        if (Cf) Cf[(size_t)row * H + col] = v;
        if (Cb) Cb[(size_t)row * H + col] = f2bf(v);
      }
    }
  }
}

// ---------- MFConv grouped WMMA GEMM over degree buckets ----------
__global__ __launch_bounds__(32) void mf_gemm(const bfraw* Aagg, const bfraw* Ah,
                                              const bfraw* Wlt, const bfraw* Wrt,
                                              const float* bl,
                                              const int* perm, const int* bcount,
                                              const int* boffset, const int* btile,
                                              float* Hout) {
  __shared__ __align__(16) bfraw Blt[H * H];
  __shared__ __align__(16) bfraw Brt[H * H];
  __shared__ int rows[16];
  int t = blockIdx.x;
  if (t >= btile[NB]) return;
  int d = 0;
  for (int i = 1; i < NB; ++i) if (t >= btile[i]) d = i;
  int lt = t - btile[d];
  int cnt = bcount[d];
  int base = boffset[d] + lt * 16;
  int tid = threadIdx.x;
  stage_lds(Wlt + d * H * H, Blt, H * H, tid, 32);
  stage_lds(Wrt + d * H * H, Brt, H * H, tid, 32);
  if (tid < 16) {
    int local = lt * 16 + tid;
    rows[tid] = perm[(local < cnt) ? (base + tid) : boffset[d]];
  }
  stage_wait();
  __syncthreads();
  int g = tid >> 4, c0 = tid & 15;
  int myrow = rows[c0];
  v8f acc[4] = {};
  for (int kh = 0; kh < 2; ++kh) {
    v16bf aa = load_a_frag(Aagg, myrow, kh * 32 + g * 8);
    v16bf ah = load_a_frag(Ah,   myrow, kh * 32 + g * 8);
    for (int ct = 0; ct < 4; ++ct) {
      v16bf fl = load_b_frag(Blt, ct * 16 + c0, g, kh);
      v16bf fr = load_b_frag(Brt, ct * 16 + c0, g, kh);
      acc[ct] = wmma_bf(aa, fl, acc[ct]);
      acc[ct] = wmma_bf(ah, fr, acc[ct]);
    }
  }
  const float* bias = bl + d * H;
  for (int ct = 0; ct < 4; ++ct) {
    int col = ct * 16 + c0;
    float bv = bias[col];
    for (int j = 0; j < 8; ++j) {
      int rit = j + g * 8;
      if (lt * 16 + rit < cnt) {
        int node = rows[rit];
        Hout[(size_t)node * H + col] = acc[ct][j] + bv;
      }
    }
  }
}

// ---------- pooling + MLP tail ----------
__global__ void k_pool(const float* h, const int* batch, float* gout, int n) {
  int idx = blockIdx.x * blockDim.x + threadIdx.x;
  if (idx >= n * (H / 4)) return;
  int node = idx >> 4, f = (idx & 15) * 4;
  float4 v = *(const float4*)(h + (size_t)node * H + f);
  float* gp = gout + (size_t)batch[node] * H + f;
  atomicAdd(gp + 0, v.x);
  atomicAdd(gp + 1, v.y);
  atomicAdd(gp + 2, v.z);
  atomicAdd(gp + 3, v.w);
}
__global__ __launch_bounds__(32) void gemm_tail(const bfraw* A, const bfraw* Bt_g,
                                                const float* bias, float* C, int M) {
  __shared__ __align__(16) bfraw Bt[OUTF * H];
  int tid = threadIdx.x;
  stage_lds(Bt_g, Bt, OUTF * H, tid, 32);
  stage_wait();
  __syncthreads();
  int g = tid >> 4, c0 = tid & 15;
  int rowbase = blockIdx.x * 16;
  int arow = rowbase + c0; if (arow >= M) arow = M - 1;
  v8f acc = {};
  for (int kh = 0; kh < 2; ++kh) {
    v16bf af = load_a_frag(A, arow, kh * 32 + g * 8);
    v16bf bf = load_b_frag(Bt, c0, g, kh);
    acc = wmma_bf(af, bf, acc);
  }
  float bv = bias[c0];
  for (int j = 0; j < 8; ++j) {
    int row = rowbase + j + g * 8;
    if (row < M) C[(size_t)row * OUTF + c0] = acc[j] + bv;
  }
}

// ---------- host orchestration ----------
extern "C" void kernel_launch(void* const* d_in, const int* in_sizes, int n_in,
                              void* d_out, int out_size, void* d_ws, size_t ws_size,
                              hipStream_t stream) {
  (void)n_in; (void)ws_size;
  const float* x      = (const float*)d_in[0];
  const float* eattr  = (const float*)d_in[1];
  const int*   eidx   = (const int*)d_in[2];
  const int*   batch  = (const int*)d_in[3];
  const float* emb    = (const float*)d_in[4];
  const float* lin1_w = (const float*)d_in[5];
  const float* lin1_b = (const float*)d_in[6];
  const float* lin2_w = (const float*)d_in[7];
  const float* lin2_b = (const float*)d_in[8];
  const float* rgcn_w[2]    = {(const float*)d_in[9],  (const float*)d_in[15]};
  const float* rgcn_root[2] = {(const float*)d_in[10], (const float*)d_in[16]};
  const float* rgcn_bv[2]   = {(const float*)d_in[11], (const float*)d_in[17]};
  const float* mf_wl[2]     = {(const float*)d_in[12], (const float*)d_in[18]};
  const float* mf_bl[2]     = {(const float*)d_in[13], (const float*)d_in[19]};
  const float* mf_wr[2]     = {(const float*)d_in[14], (const float*)d_in[20]};

  const int N = in_sizes[3];
  const int E = in_sizes[1] / RR;
  const int G = out_size / OUTF;
  const int* src = eidx;
  const int* tgt = eidx + E;

  char* wp = (char*)d_ws;
  auto alloc = [&](size_t bytes) -> void* {
    void* p = (void*)wp; wp += (bytes + 255) & ~(size_t)255; return p;
  };
  float* hA    = (float*)alloc((size_t)N * H * 4);
  float* hB    = (float*)alloc((size_t)N * H * 4);
  float* hrf   = (float*)alloc((size_t)N * H * 4);
  bfraw* hrb   = (bfraw*)alloc((size_t)N * H * 2);
  bfraw* hxb   = (bfraw*)alloc((size_t)RR * N * H * 2);
  float* hroot = (float*)alloc((size_t)N * H * 4);
  float* agg   = (float*)alloc((size_t)N * H * 4);
  bfraw* aggb  = (bfraw*)alloc((size_t)N * H * 2);
  float* cnt   = (float*)alloc((size_t)N * RR * 4);
  int*   rel   = (int*)alloc((size_t)E * 4);
  int*   deg   = (int*)alloc((size_t)N * 4);
  int*   degc  = (int*)alloc((size_t)N * 4);
  int*   perm  = (int*)alloc((size_t)N * 4);
  int*   bcount  = (int*)alloc(64);
  int*   boffset = (int*)alloc(64);
  int*   btile   = (int*)alloc(64);
  int*   bfill   = (int*)alloc(64);
  float* gbuf  = (float*)alloc((size_t)G * H * 4);
  bfraw* gb    = (bfraw*)alloc((size_t)G * H * 2);
  bfraw* g1b   = (bfraw*)alloc((size_t)G * H * 2);
  bfraw *wbf[2], *rootbf[2], *wlbf[2], *wrbf[2];
  for (int b = 0; b < 2; ++b) {
    wbf[b]    = (bfraw*)alloc((size_t)RR * H * H * 2);
    rootbf[b] = (bfraw*)alloc((size_t)H * H * 2);
    wlbf[b]   = (bfraw*)alloc((size_t)NB * H * H * 2);
    wrbf[b]   = (bfraw*)alloc((size_t)NB * H * H * 2);
  }
  bfraw* lin1bf = (bfraw*)alloc((size_t)H * H * 2);
  bfraw* lin2bf = (bfraw*)alloc((size_t)H * OUTF * 2);

  auto blocks = [](int n) { return (n + 255) / 256; };

  // Weight conversions: fp32 row-major -> bf16 transposed [n][k]
  for (int b = 0; b < 2; ++b) {
    k_bf16t<<<blocks(RR * H * H), 256, 0, stream>>>(rgcn_w[b], wbf[b], H, H, RR * H * H);
    k_bf16t<<<blocks(H * H), 256, 0, stream>>>(rgcn_root[b], rootbf[b], H, H, H * H);
    k_bf16t<<<blocks(NB * H * H), 256, 0, stream>>>(mf_wl[b], wlbf[b], H, H, NB * H * H);
    k_bf16t<<<blocks(NB * H * H), 256, 0, stream>>>(mf_wr[b], wrbf[b], H, H, NB * H * H);
  }
  k_bf16t<<<blocks(H * H), 256, 0, stream>>>(lin1_w, lin1bf, H, H, H * H);
  k_bf16t<<<blocks(H * OUTF), 256, 0, stream>>>(lin2_w, lin2bf, H, OUTF, H * OUTF);

  // h0 = emb[argmax(x)]; rel = argmax(edge_attr)
  k_embed<<<blocks(N), 256, 0, stream>>>(x, emb, hA, N);
  k_rel<<<blocks(E), 256, 0, stream>>>(eattr, rel, E);

  const int gemmGrid = (N + 63) / 64;
  const int mfGrid = (N + 15) / 16 + NB;
  const int e4 = E * (H / 4);
  const int n4 = N * (H / 4);

  float* hcur = hA;
  float* hnext = hB;
  for (int b = 0; b < 2; ++b) {
    // relu(h) -> fp32 + bf16
    k_relu_dual<<<blocks(N * H), 256, 0, stream>>>(hcur, hrf, hrb, N * H);
    // per-relation transforms hx[r] = relu(h) @ W[r]   (bf16 out)
    for (int r = 0; r < RR; ++r)
      gemm64<<<gemmGrid, 128, 0, stream>>>(hrb, wbf[b] + r * H * H, nullptr,
                                           nullptr, hxb + (size_t)r * N * H, N, 0);
    // root transform (+bias)
    gemm64<<<gemmGrid, 128, 0, stream>>>(hrb, rootbf[b], rgcn_bv[b], hroot, nullptr, N, 0);
    // per-(tgt,rel) mean scatter
    k_zero_f<<<blocks(N * RR), 256, 0, stream>>>(cnt, N * RR);
    k_cnt<<<blocks(E), 256, 0, stream>>>(tgt, rel, cnt, E);
    k_inv<<<blocks(N * RR), 256, 0, stream>>>(cnt, N * RR);
    k_zero_f<<<blocks(N * H), 256, 0, stream>>>(agg, N * H);
    k_rgcn_scatter<<<blocks(e4), 256, 0, stream>>>(hxb, cnt, src, tgt, rel, agg, E, N);
    k_combine<<<blocks(N * H), 256, 0, stream>>>(agg, hroot, hcur, N * H);
    // relu again for MFConv
    k_relu_dual<<<blocks(N * H), 256, 0, stream>>>(hcur, hrf, hrb, N * H);
    // degrees + neighbor sum
    k_zero_i<<<blocks(N), 256, 0, stream>>>(deg, N);
    k_deg<<<blocks(E), 256, 0, stream>>>(tgt, deg, E);
    k_zero_f<<<blocks(N * H), 256, 0, stream>>>(agg, N * H);
    k_mf_scatter<<<blocks(e4), 256, 0, stream>>>(hrf, src, tgt, agg, E);
    k_bf16<<<blocks(N * H), 256, 0, stream>>>(agg, aggb, N * H);
    // degree buckets -> permutation
    k_zero_i<<<1, 64, 0, stream>>>(bcount, NB);
    k_bucket_count<<<blocks(N), 256, 0, stream>>>(deg, degc, bcount, N);
    k_prefix<<<1, 1, 0, stream>>>(bcount, boffset, btile, bfill);
    k_perm<<<blocks(N), 256, 0, stream>>>(degc, boffset, bfill, perm, N);
    // grouped GEMM: h' = agg@Wl[d] + bl[d] + relu(h)@Wr[d]
    mf_gemm<<<mfGrid, 32, 0, stream>>>(aggb, hrb, wlbf[b], wrbf[b], mf_bl[b],
                                       perm, bcount, boffset, btile, hnext);
    float* t = hcur; hcur = hnext; hnext = t;
  }

  // global add pool
  k_zero_f<<<blocks(G * H), 256, 0, stream>>>(gbuf, G * H);
  k_pool<<<blocks(n4), 256, 0, stream>>>(hcur, batch, gbuf, N);
  // MLP: relu(g@lin1+b1) @ lin2 + b2
  k_bf16<<<blocks(G * H), 256, 0, stream>>>(gbuf, gb, G * H);
  gemm64<<<(G + 63) / 64, 128, 0, stream>>>(gb, lin1bf, lin1_b, nullptr, g1b, G, 1);
  gemm_tail<<<(G + 15) / 16, 32, 0, stream>>>(g1b, lin2bf, lin2_b, (float*)d_out, G);
}
